// MoELayer_15461882265904
// MI455X (gfx1250) — matrix operand order repaired and checked by reading, not compiled, not run on hardware
//
#include <hip/hip_runtime.h>
#include <hip/hip_bf16.h>
#include <stdint.h>

// Problem constants (from reference)
#define DIMX 1024
#define HIDX 2048
#define NEXP 8
#define NTOK 1024          // B*T = 2*512
#define TILE_M 32

typedef __attribute__((ext_vector_type(16))) __bf16       v16bf;
typedef __attribute__((ext_vector_type(8)))  float        v8f;
typedef __attribute__((ext_vector_type(8)))  unsigned int v8u;

// Build a 16-element bf16 WMMA operand from two 16-byte chunks.
static __device__ inline v16bf load_bf16x16(const __bf16* p0, const __bf16* p1) {
    uint4 a = *reinterpret_cast<const uint4*>(p0);
    uint4 b = *reinterpret_cast<const uint4*>(p1);
    v8u v;
    v[0] = a.x; v[1] = a.y; v[2] = a.z; v[3] = a.w;
    v[4] = b.x; v[5] = b.y; v[6] = b.z; v[7] = b.w;
    return __builtin_bit_cast(v16bf, v);
}

// ---------------------------------------------------------------------------
// fp32 -> bf16 conversion (vectorized, grid-stride)
// ---------------------------------------------------------------------------
__global__ void cvt_f32_bf16(const float* __restrict__ src,
                             __bf16* __restrict__ dst, int n4) {
    int i = blockIdx.x * blockDim.x + threadIdx.x;
    int stride = gridDim.x * blockDim.x;
    for (; i < n4; i += stride) {
        float4 f = reinterpret_cast<const float4*>(src)[i];
        __bf16 b0 = (__bf16)f.x, b1 = (__bf16)f.y, b2 = (__bf16)f.z, b3 = (__bf16)f.w;
        ushort4 p;
        p.x = __builtin_bit_cast(unsigned short, b0);
        p.y = __builtin_bit_cast(unsigned short, b1);
        p.z = __builtin_bit_cast(unsigned short, b2);
        p.w = __builtin_bit_cast(unsigned short, b3);
        reinterpret_cast<ushort4*>(dst)[i] = p;
    }
}

// ---------------------------------------------------------------------------
// Router: one wave32 per token. logits = x @ router_w^T, softmax, top-2,
// renormalize over the 2 selected -> combine[N][E] (0 for non-selected).
// ---------------------------------------------------------------------------
__global__ __launch_bounds__(256, 1)
void router_kernel(const float* __restrict__ x, const float* __restrict__ rw,
                   float* __restrict__ combine) {
    const int wave = threadIdx.x >> 5;
    const int lane = threadIdx.x & 31;
    const int n = blockIdx.x * 8 + wave;      // 8 waves per block
    if (n >= NTOK) return;
    const float* xr = x + (size_t)n * DIMX;
    float logits[NEXP];
#pragma unroll
    for (int e = 0; e < NEXP; ++e) {
        const float* w = rw + (size_t)e * DIMX;
        float s = 0.f;
        for (int i = lane; i < DIMX; i += 32) s += xr[i] * w[i];
#pragma unroll
        for (int o = 16; o > 0; o >>= 1) s += __shfl_xor(s, o, 32);
        logits[e] = s;
    }
    if (lane == 0) {
        int i0 = 0;
#pragma unroll
        for (int e = 1; e < NEXP; ++e) if (logits[e] > logits[i0]) i0 = e;
        int i1 = -1;
#pragma unroll
        for (int e = 0; e < NEXP; ++e) {
            if (e == i0) continue;
            if (i1 < 0 || logits[e] > logits[i1]) i1 = e;
        }
        // softmax over top-2 (global softmax denominators cancel on renorm)
        float w1 = __expf(logits[i1] - logits[i0]);
        float inv = 1.0f / (1.0f + w1);
        float* c = combine + (size_t)n * NEXP;
#pragma unroll
        for (int e = 0; e < NEXP; ++e) c[e] = 0.f;
        c[i0] = inv;
        c[i1] = w1 * inv;
    }
}

// ---------------------------------------------------------------------------
// Fused expert kernel: grid = (NTOK/TILE_M, NEXP), 256 threads (8 waves).
//   GEMM1: G,U (32 x 2048, K=1024) via v_wmma_f32_16x16x32_bf16, wave owns a
//          256-wide HID slice.  h = silu(g)*u -> bf16 in LDS.
//   GEMM2: Y (32 x 1024, K=2048), A=h from LDS, B=w_down rows from L2.
//   out[n,d] += combine[n,e] * Y   (f32 atomics; out pre-zeroed)
// LDS: sX 32x(1024+8) bf16 = 66,048 B ; sH 32x(2048+8) bf16 = 131,584 B
// ---------------------------------------------------------------------------
#define SX_STRIDE (DIMX + 8)
#define SH_STRIDE (HIDX + 8)

__global__ __launch_bounds__(256, 1)
void moe_expert_kernel(const __bf16* __restrict__ xb,
                       const __bf16* __restrict__ gb,
                       const __bf16* __restrict__ ub,
                       const __bf16* __restrict__ db,
                       const float* __restrict__ combine,
                       float* __restrict__ out) {
    __shared__ __bf16 sX[TILE_M * SX_STRIDE];
    __shared__ __bf16 sH[TILE_M * SH_STRIDE];

    const int e      = blockIdx.y;
    const int m0     = blockIdx.x * TILE_M;
    const int tid    = threadIdx.x;
    const int lane   = tid & 31;
    const int wave   = tid >> 5;
    const int laneHi = lane >> 4;     // 0: lanes 0-15, 1: lanes 16-31
    const int lane15 = lane & 15;

    // ---- stage x tile (32 x 1024 bf16) into LDS with padded rows ----
    {
        const uint4* src = reinterpret_cast<const uint4*>(xb + (size_t)m0 * DIMX);
        uint4* dst = reinterpret_cast<uint4*>(sX);
        // 32 rows * 128 uint4/row ; padded LDS stride = 129 uint4
#pragma unroll
        for (int i = tid; i < TILE_M * (DIMX / 8); i += 256) {
            int r = i >> 7;
            int c = i & 127;
            dst[r * 129 + c] = src[r * 128 + c];
        }
    }
    __syncthreads();

    // ---- GEMM1: gate & up for this wave's HID slice ----
    const int SLICE = HIDX / 8;             // 256 hid columns per wave
    const int hbase = wave * SLICE;
    const __bf16* wg_e = gb + (size_t)e * HIDX * DIMX;
    const __bf16* wu_e = ub + (size_t)e * HIDX * DIMX;

    for (int ht = 0; ht < SLICE; ht += 16) {
        const int ncol = hbase + ht + lane15;           // B column this lane holds
        const __bf16* bgrow = wg_e + (size_t)ncol * DIMX;
        const __bf16* burow = wu_e + (size_t)ncol * DIMX;
        const __bf16* a0row = sX + (size_t)lane15 * SX_STRIDE;         // A tile 0 (M 0..15)
        const __bf16* a1row = sX + (size_t)(16 + lane15) * SX_STRIDE;  // A tile 1 (M 16..31)

        v8f accg0 = {}, accg1 = {}, accu0 = {}, accu1 = {};
#pragma unroll 4
        for (int k = 0; k < DIMX; k += 32) {
            const int kb = k + 16 * laneHi;             // B: contiguous 16 elems per lane
            v16bf Bg = load_bf16x16(bgrow + kb, bgrow + kb + 8);
            v16bf Bu = load_bf16x16(burow + kb, burow + kb + 8);
            const int ka = k + 16 * laneHi;             // A: contiguous 16 elems per lane
            v16bf A0 = load_bf16x16(a0row + ka, a0row + ka + 8);
            v16bf A1 = load_bf16x16(a1row + ka, a1row + ka + 8);
            accg0 = __builtin_amdgcn_wmma_f32_16x16x32_bf16(false, A0, false, Bg,
                                                            (short)0, accg0, false, false);
            accg1 = __builtin_amdgcn_wmma_f32_16x16x32_bf16(false, A1, false, Bg,
                                                            (short)0, accg1, false, false);
            accu0 = __builtin_amdgcn_wmma_f32_16x16x32_bf16(false, A0, false, Bu,
                                                            (short)0, accu0, false, false);
            accu1 = __builtin_amdgcn_wmma_f32_16x16x32_bf16(false, A1, false, Bu,
                                                            (short)0, accu1, false, false);
        }
        // h = silu(g) * u -> LDS (bf16).  C layout: VGPR r, lanes<16 -> M=r,
        // lanes>=16 -> M=r+8, N=lane15.
        const int col = hbase + ht + lane15;
#pragma unroll
        for (int r = 0; r < 8; ++r) {
            int mr = r + 8 * laneHi;
            float g0 = accg0[r];
            float h0 = (g0 / (1.0f + __expf(-g0))) * accu0[r];
            sH[(size_t)mr * SH_STRIDE + col] = (__bf16)h0;
            float g1 = accg1[r];
            float h1 = (g1 / (1.0f + __expf(-g1))) * accu1[r];
            sH[(size_t)(16 + mr) * SH_STRIDE + col] = (__bf16)h1;
        }
    }
    __syncthreads();

    // ---- GEMM2: y = h @ w_down^T ; scale by combine; atomic add to out ----
    const __bf16* wd_e = db + (size_t)e * DIMX * HIDX;
    const int dbase = wave * (DIMX / 8);                // 128 out columns per wave
    for (int dt = 0; dt < DIMX / 8; dt += 16) {
        const int dcol = dbase + dt + lane15;
        const __bf16* bdrow = wd_e + (size_t)dcol * HIDX;
        const __bf16* a0row = sH + (size_t)lane15 * SH_STRIDE;
        const __bf16* a1row = sH + (size_t)(16 + lane15) * SH_STRIDE;

        v8f acc0 = {}, acc1 = {};
#pragma unroll 4
        for (int k = 0; k < HIDX; k += 32) {
            const int kb = k + 16 * laneHi;
            v16bf Bd = load_bf16x16(bdrow + kb, bdrow + kb + 8);
            const int ka = k + 16 * laneHi;
            v16bf A0 = load_bf16x16(a0row + ka, a0row + ka + 8);
            v16bf A1 = load_bf16x16(a1row + ka, a1row + ka + 8);
            acc0 = __builtin_amdgcn_wmma_f32_16x16x32_bf16(false, A0, false, Bd,
                                                           (short)0, acc0, false, false);
            acc1 = __builtin_amdgcn_wmma_f32_16x16x32_bf16(false, A1, false, Bd,
                                                           (short)0, acc1, false, false);
        }
#pragma unroll
        for (int r = 0; r < 8; ++r) {
            int mr = r + 8 * laneHi;
            float c0 = combine[(size_t)(m0 + mr) * NEXP + e];
            atomicAdd(&out[(size_t)(m0 + mr) * DIMX + dcol], acc0[r] * c0);
            int mr1 = 16 + mr;
            float c1 = combine[(size_t)(m0 + mr1) * NEXP + e];
            atomicAdd(&out[(size_t)(m0 + mr1) * DIMX + dcol], acc1[r] * c1);
        }
    }
}

// ---------------------------------------------------------------------------
// Host launcher.  Workspace layout (bytes):
//   [0)            x_bf16      : NTOK*DIMX*2          =  2,097,152
//   [xb)           w_gate bf16 : NEXP*HIDX*DIMX*2     = 33,554,432
//   [+W)           w_up   bf16 : 33,554,432
//   [+W)           w_down bf16 : 33,554,432
//   [+W)           combine f32 : NTOK*NEXP*4          =     32,768
//   total ~ 98 MB
// ---------------------------------------------------------------------------
extern "C" void kernel_launch(void* const* d_in, const int* in_sizes, int n_in,
                              void* d_out, int out_size, void* d_ws, size_t ws_size,
                              hipStream_t stream) {
    const float* x        = (const float*)d_in[0];
    const float* router_w = (const float*)d_in[1];
    const float* w_gate   = (const float*)d_in[2];
    const float* w_up     = (const float*)d_in[3];
    const float* w_down   = (const float*)d_in[4];
    float* out = (float*)d_out;

    const size_t XB = (size_t)NTOK * DIMX * 2;
    const size_t WB = (size_t)NEXP * HIDX * DIMX * 2;
    char* ws = (char*)d_ws;
    __bf16* xb = (__bf16*)ws;
    __bf16* gb = (__bf16*)(ws + XB);
    __bf16* ub = (__bf16*)(ws + XB + WB);
    __bf16* db = (__bf16*)(ws + XB + 2 * WB);
    float* combine = (float*)(ws + XB + 3 * WB);

    // zero the output (we accumulate with f32 atomics)
    hipMemsetAsync(out, 0, sizeof(float) * (size_t)NTOK * DIMX, stream);

    // fp32 -> bf16 staging (weights then fit in L2 as bf16)
    cvt_f32_bf16<<<1024, 256, 0, stream>>>(x, xb, NTOK * DIMX / 4);
    cvt_f32_bf16<<<2048, 256, 0, stream>>>(w_gate, gb, NEXP * HIDX * DIMX / 4);
    cvt_f32_bf16<<<2048, 256, 0, stream>>>(w_up, ub, NEXP * HIDX * DIMX / 4);
    cvt_f32_bf16<<<2048, 256, 0, stream>>>(w_down, db, NEXP * HIDX * DIMX / 4);

    // router -> combine weights
    router_kernel<<<NTOK / 8, 256, 0, stream>>>(x, router_w, combine);

    // fused expert compute
    dim3 grid(NTOK / TILE_M, NEXP);
    moe_expert_kernel<<<grid, 256, 0, stream>>>(xb, gb, ub, db, combine, out);
}
